// NodeAttDiff_58884001628629
// MI455X (gfx1250) — compile-verified, matrix operand order checked by reading, not run on hardware
//
#include <hip/hip_runtime.h>
#include <hip/hip_bf16.h>

typedef __attribute__((ext_vector_type(4)))  __bf16 v4bf;
typedef __attribute__((ext_vector_type(8)))  __bf16 v8bf;
typedef __attribute__((ext_vector_type(16))) __bf16 v16bf;
typedef __attribute__((ext_vector_type(8)))  float  v8f;

#define Dm 256          // feature dim
#define K3 768          // 3*D concat K
#define KF (K3 / 32)    // 24 K-fragments
#define MT 128          // rows per block in GEMM kernel
#define APITCH 776      // 768 + 8 bf16 pad -> row stride 1552B = 388 dwords (bank rotate)
#define NB 32           // N columns staged per slab round (2 WMMA tiles)

// ---------------------------------------------------------------------------
// Kernel 0: convert + transpose W1 (f32 [768][256]) -> bf16 W1T [256][768]
// ---------------------------------------------------------------------------
__global__ void prep_w1t_kernel(const float* __restrict__ W1, __bf16* __restrict__ w1t) {
    int idx = blockIdx.x * 256 + threadIdx.x;     // 768*256 total
    int k = idx >> 8;
    int n = idx & 255;
    w1t[n * K3 + k] = (__bf16)W1[idx];
}

// ---------------------------------------------------------------------------
// fragment load: two 16B LDS reads -> 16 bf16 lane fragment
// ---------------------------------------------------------------------------
__device__ __forceinline__ v16bf ldfrag(const __bf16* p) {
    v8bf lo = *(const v8bf*)(p);
    v8bf hi = *(const v8bf*)(p + 16);
    return __builtin_shufflevector(lo, hi, 0,1,2,3,4,5,6,7,8,9,10,11,12,13,14,15);
}

// ---------------------------------------------------------------------------
// Kernel 1: fused  raw = relu([e1|e2|e1-e2] @ W1 + b1) @ W2   (b2 cancels)
// 256 threads = 8 waves; wave w owns rows [blk*128 + 16w, +16)
// A tile register-resident (24 frags); 2 N-tiles per wave; B loads pipelined;
// W slab staged via async global->LDS DMA
// ---------------------------------------------------------------------------
__global__ void __launch_bounds__(256, 1)
fused_mlp_raw_kernel(const float* __restrict__ out_gnn,
                     const __bf16* __restrict__ w1t,
                     const float* __restrict__ b1,
                     const float* __restrict__ w2,
                     float* __restrict__ raw,
                     int half) {
    extern __shared__ char smem_raw_[];
    __bf16* As = (__bf16*)smem_raw_;            // MT x APITCH   (~194 KB)
    __bf16* Bs = As + MT * APITCH;              // NB x APITCH   (~48.5 KB)

    const int tid  = threadIdx.x;
    const int row0 = blockIdx.x * MT;

    // ---- stage A tile: bf16 [e1 | e2 | diff], float4 loads, b64 LDS stores ----
    for (int idx = tid; idx < MT * (Dm / 4); idx += 256) {
        int r  = idx >> 6;            // / 64 quads per row
        int c4 = (idx & 63) << 2;
        int gr = row0 + r;
        float4 va = make_float4(0.f, 0.f, 0.f, 0.f);
        float4 vb = make_float4(0.f, 0.f, 0.f, 0.f);
        if (gr < half) {
            va = *(const float4*)(out_gnn + (size_t)gr * Dm + c4);
            vb = *(const float4*)(out_gnn + (size_t)(gr + half) * Dm + c4);
        }
        __bf16* Arow = As + r * APITCH;
        v4bf pa = {(__bf16)va.x, (__bf16)va.y, (__bf16)va.z, (__bf16)va.w};
        v4bf pb = {(__bf16)vb.x, (__bf16)vb.y, (__bf16)vb.z, (__bf16)vb.w};
        v4bf pd = {(__bf16)(va.x - vb.x), (__bf16)(va.y - vb.y),
                   (__bf16)(va.z - vb.z), (__bf16)(va.w - vb.w)};
        *(v4bf*)(Arow + c4)          = pa;
        *(v4bf*)(Arow + Dm + c4)     = pb;
        *(v4bf*)(Arow + 2 * Dm + c4) = pd;
    }
    __syncthreads();

    const int wave  = tid >> 5;
    const int lane  = tid & 31;
    const int lhalf = lane >> 4;    // 0: K +0..7/+16..23   1: K +8..15/+24..31
    const int l16   = lane & 15;

    const __bf16* Abase  = As + (wave * 16 + l16) * APITCH + lhalf * 8;
    const __bf16* Bbase0 = Bs + l16 * APITCH + lhalf * 8;
    const __bf16* Bbase1 = Bbase0 + 16 * APITCH;

    // ---- hoist the wave's whole A tile into registers (24 x 8 VGPRs) ----
    v16bf afrag[KF];
#pragma unroll
    for (int kk = 0; kk < KF; ++kk)
        afrag[kk] = ldfrag(Abase + kk * 32);

    float rawacc[8];
#pragma unroll
    for (int v = 0; v < 8; ++v) rawacc[v] = 0.f;

    const unsigned bs_lds = (unsigned)(uintptr_t)Bs;   // low 32 bits = LDS offset

    for (int nc2 = 0; nc2 < Dm / NB; ++nc2) {
        __syncthreads();   // previous round done reading Bs
        // ---- async DMA: NB x 768 slab of W1T -> LDS (rows contiguous, b128) ----
        {
            unsigned long long gbase =
                (unsigned long long)(uintptr_t)(w1t + (size_t)(nc2 * NB) * K3);
            for (int idx = tid; idx < NB * 96; idx += 256) {
                int r = idx / 96;
                int c = idx - r * 96;
                unsigned ldsoff = bs_lds + (unsigned)(r * (APITCH * 2) + c * 16);
                unsigned voff   = (unsigned)(idx * 16);
                asm volatile("global_load_async_to_lds_b128 %0, %1, %2"
                             :: "v"(ldsoff), "v"(voff), "s"(gbase)
                             : "memory");
            }
            asm volatile("s_wait_asynccnt 0" ::: "memory");
        }
        __syncthreads();

        v8f acc0 = {};
        v8f acc1 = {};
        // ---- pipelined B fragments over K (A already in registers) ----
        v16bf b0  = ldfrag(Bbase0);
        v16bf b1f = ldfrag(Bbase1);
#pragma unroll
        for (int kk = 0; kk < KF; ++kk) {
            v16bf b0n = b0, b1n = b1f;
            if (kk + 1 < KF) {
                b0n = ldfrag(Bbase0 + (kk + 1) * 32);
                b1n = ldfrag(Bbase1 + (kk + 1) * 32);
            }
            acc0 = __builtin_amdgcn_wmma_f32_16x16x32_bf16(
                false, afrag[kk], false, b0, (short)0, acc0, false, false);
            acc1 = __builtin_amdgcn_wmma_f32_16x16x32_bf16(
                false, afrag[kk], false, b1f, (short)0, acc1, false, false);
            b0 = b0n; b1f = b1n;
        }

        // ---- fused epilogue: +b1, relu, * W2 column weight (both N-tiles) ----
        int n0 = nc2 * NB + l16;               // C layout: N = lane%16
        float b1v0 = b1[n0],      w2v0 = w2[n0];
        float b1v1 = b1[n0 + 16], w2v1 = w2[n0 + 16];
#pragma unroll
        for (int v = 0; v < 8; ++v) {          // C layout: M = v + 8*lhalf
            float h0 = fmaxf(acc0[v] + b1v0, 0.f);
            float h1 = fmaxf(acc1[v] + b1v1, 0.f);
            rawacc[v] += h0 * w2v0 + h1 * w2v1;
        }
    }

    // ---- reduce over the 16 N-lanes of each half, store raw ----
#pragma unroll
    for (int v = 0; v < 8; ++v) {
        float s = rawacc[v];
        s += __shfl_xor(s, 1, 32);
        s += __shfl_xor(s, 2, 32);
        s += __shfl_xor(s, 4, 32);
        s += __shfl_xor(s, 8, 32);
        rawacc[v] = s;
    }
    if (l16 == 0) {
        int gr = row0 + wave * 16 + lhalf * 8;
#pragma unroll
        for (int v = 0; v < 8; ++v)
            if (gr + v < half) raw[gr + v] = rawacc[v];
    }
}

// ---------------------------------------------------------------------------
// Kernel 2: per-segment softmax + weighted diff reduction (batch is sorted)
// one block per segment, thread t owns output column t
// ---------------------------------------------------------------------------
__device__ __forceinline__ int lbound(const int* __restrict__ b, int n, int key) {
    int lo = 0, hi = n;
    while (lo < hi) {
        int mid = (lo + hi) >> 1;
        if (b[mid] < key) lo = mid + 1; else hi = mid;
    }
    return lo;
}

__global__ void __launch_bounds__(256)
segment_att_kernel(const float* __restrict__ out_gnn,
                   const int* __restrict__ batch,
                   const float* __restrict__ raw,
                   float* __restrict__ out,
                   int half) {
    const int s   = blockIdx.x;
    const int tid = threadIdx.x;
    const int lo  = lbound(batch, half, s);
    const int hi  = lbound(batch, half, s + 1);

    if (lo >= hi) { out[(size_t)s * Dm + tid] = 0.f; return; }

    __shared__ float red[8];
    __shared__ float sbc[2];        // [0]=max, [1]=sum
    __shared__ float wlds[256];

    const int wave = tid >> 5;
    const int lane = tid & 31;

    // ---- pass 1: segment max ----
    float m = -INFINITY;
    for (int i = lo + tid; i < hi; i += 256) m = fmaxf(m, raw[i]);
    m = fmaxf(m, __shfl_xor(m, 16, 32));
    m = fmaxf(m, __shfl_xor(m, 8, 32));
    m = fmaxf(m, __shfl_xor(m, 4, 32));
    m = fmaxf(m, __shfl_xor(m, 2, 32));
    m = fmaxf(m, __shfl_xor(m, 1, 32));
    if (lane == 0) red[wave] = m;
    __syncthreads();
    if (tid == 0) {
        float mm = red[0];
#pragma unroll
        for (int w = 1; w < 8; ++w) mm = fmaxf(mm, red[w]);
        sbc[0] = mm;
    }
    __syncthreads();
    m = sbc[0];

    // ---- pass 2: sum of exp ----
    float se = 0.f;
    for (int i = lo + tid; i < hi; i += 256) se += __expf(raw[i] - m);
    se += __shfl_xor(se, 16, 32);
    se += __shfl_xor(se, 8, 32);
    se += __shfl_xor(se, 4, 32);
    se += __shfl_xor(se, 2, 32);
    se += __shfl_xor(se, 1, 32);
    if (lane == 0) red[wave] = se;
    __syncthreads();
    if (tid == 0) {
        float ss = red[0];
#pragma unroll
        for (int w = 1; w < 8; ++w) ss += red[w];
        sbc[1] = ss;
    }
    __syncthreads();
    const float inv = 1.0f / sbc[1];

    // ---- pass 3: out[s][c] = sum_i att_i * (e1[i][c] - e2[i][c]) ----
    float acc = 0.f;
    for (int base = lo; base < hi; base += 256) {
        __syncthreads();
        int i = base + tid;
        wlds[tid] = (i < hi) ? __expf(raw[i] - m) * inv : 0.f;
        __syncthreads();
        int cnt = min(256, hi - base);
#pragma unroll 4
        for (int j = 0; j < cnt; ++j) {
            size_t r = (size_t)(base + j);
            float w = wlds[j];
            float d = out_gnn[r * Dm + tid] - out_gnn[(r + half) * Dm + tid];
            acc += w * d;
        }
    }
    out[(size_t)s * Dm + tid] = acc;
}

// ---------------------------------------------------------------------------
extern "C" void kernel_launch(void* const* d_in, const int* in_sizes, int n_in,
                              void* d_out, int out_size, void* d_ws, size_t ws_size,
                              hipStream_t stream) {
    const float* out_gnn = (const float*)d_in[0];
    const int*   batch   = (const int*)d_in[1];
    const float* W1      = (const float*)d_in[2];
    const float* b1      = (const float*)d_in[3];
    const float* W2      = (const float*)d_in[4];
    // d_in[5] = b2 : cancels inside segment softmax, unused

    const int N2   = in_sizes[0] / Dm;   // 400000
    const int half = N2 / 2;             // 200000

    __bf16* w1t = (__bf16*)d_ws;                         // 768*256*2 = 393216 B
    float*  raw = (float*)((char*)d_ws + (size_t)K3 * Dm * 2);

    // 0) W1 -> bf16 transposed
    prep_w1t_kernel<<<K3, 256, 0, stream>>>(W1, w1t);

    // 1) fused MLP -> raw scores
    const int nblk = (half + MT - 1) / MT;
    const size_t smem = (size_t)(MT + NB) * APITCH * sizeof(__bf16);  // ~248 KB of 320 KB WGP LDS
    fused_mlp_raw_kernel<<<nblk, 256, smem, stream>>>(out_gnn, w1t, b1, W2, raw, half);

    // 2) segmented softmax + weighted reduction
    segment_att_kernel<<<512, 256, 0, stream>>>(out_gnn, batch, raw, (float*)d_out, half);
}
